// ContextGeometricPriorModule_88390426951952
// MI455X (gfx1250) — matrix operand order repaired and checked by reading, not compiled, not run on hardware
//
#include <hip/hip_runtime.h>
#include <hip/hip_bf16.h>

typedef __attribute__((ext_vector_type(16))) _Float16 v16h;
typedef __attribute__((ext_vector_type(8)))  float    v8f;
typedef __attribute__((ext_vector_type(2)))  float    v2f;

namespace {

constexpr int B_ = 2, T_ = 8, C_ = 64, R_ = 16, P_ = 64, IMG_ = 64;
constexpr int ROI_ = 256, TOK_ = 128, RGBE_ = 32;
constexpr int NPIX = R_ * R_;             // 256
constexpr int Q_ = T_ * NPIX;             // 2048
constexpr int K_ = C_ * NPIX;             // 16384
constexpr int CIN_T = ROI_ + RGBE_ + 3;   // 291
constexpr int CIN_C = ROI_ + RGBE_ + 12;  // 300
constexpr int PRIN = 2 * TOK_ + 5;        // 261
constexpr int PRINP = 264;                // padded per-tap channel count

// ---- output layout (floats, concatenated in reference return order) ----
constexpr size_t OUT_RET   = 0;
constexpr size_t OUT_XYZ   = OUT_RET  + (size_t)B_*T_*ROI_*NPIX;
constexpr size_t OUT_CONF  = OUT_XYZ  + (size_t)B_*T_*3*P_*P_;
constexpr size_t OUT_AMB   = OUT_CONF + (size_t)B_*T_*P_*P_;
constexpr size_t OUT_W     = OUT_AMB  + (size_t)B_*T_*P_*P_;       // weights [B,2048,16384]
constexpr size_t OUT_BXYZ  = OUT_W    + (size_t)B_*Q_*K_;
constexpr size_t OUT_BCONF = OUT_BXYZ + (size_t)B_*K_*3;

__device__ inline v8f wmma_f16(v16h a, v16h b, v8f c) {
  return __builtin_amdgcn_wmma_f32_16x16x32_f16(false, a, false, b, (short)0, c, false, false);
}
__device__ inline v8f wmma_f32x4(v2f a, v2f b, v8f c) {
  return __builtin_amdgcn_wmma_f32_16x16x4_f32(false, a, false, b, (short)0, c, false, false);
}

// =====================================================================
// K0: transpose weight matrices to [k][oc] for lane-coalesced B loads
// =====================================================================
__global__ void k_wprep(const float* __restrict__ tgt_w, const float* __restrict__ ctx_w,
                        const float* __restrict__ ret_w, const float* __restrict__ p1_w,
                        float* __restrict__ tgt_wt, float* __restrict__ ctx_wt,
                        float* __restrict__ ret_wt, float* __restrict__ p1_wt) {
  int idx = blockIdx.x * 256 + threadIdx.x;           // 0 .. 9*264*128-1
  if (idx < 292 * TOK_) {
    int c = idx / TOK_, oc = idx % TOK_;
    tgt_wt[idx] = (c < CIN_T) ? tgt_w[(size_t)oc * CIN_T + c] : 0.f;
  }
  if (idx < CIN_C * TOK_) {
    int c = idx / TOK_, oc = idx % TOK_;
    ctx_wt[idx] = ctx_w[(size_t)oc * CIN_C + c];
  }
  if (idx < TOK_ * ROI_) {
    int k = idx / ROI_, oc = idx % ROI_;
    ret_wt[idx] = ret_w[(size_t)oc * TOK_ + k];
  }
  if (idx < 9 * PRINP * TOK_) {
    int t9 = idx / (PRINP * TOK_);
    int r  = idx % (PRINP * TOK_);
    int c = r / TOK_, oc = r % TOK_;
    p1_wt[idx] = (c < PRIN) ? p1_w[((size_t)oc * PRIN + c) * 9 + t9] : 0.f;
  }
}

// =====================================================================
// K1: fused 4x4 avg-pool + 1x1 conv (3 -> 32) for target & context RGB
// =====================================================================
__global__ void k_rgbfeat(const float* __restrict__ trgb, const float* __restrict__ crgb,
                          const float* __restrict__ w, const float* __restrict__ bias,
                          float* __restrict__ tout, float* __restrict__ cout) {
  int img = blockIdx.x;            // 0..15 target, 16..143 context
  int p = threadIdx.x;             // 256 output pixels
  int py = p >> 4, px = p & 15;
  const float* src; float* dst;
  if (img < B_ * T_) { src = trgb + (size_t)img * 3 * IMG_ * IMG_; dst = tout + (size_t)img * RGBE_ * NPIX; }
  else { int j = img - B_ * T_;   src = crgb + (size_t)j   * 3 * IMG_ * IMG_; dst = cout + (size_t)j   * RGBE_ * NPIX; }
  float pool[3];
#pragma unroll
  for (int c = 0; c < 3; ++c) {
    float s = 0.f;
#pragma unroll
    for (int dy = 0; dy < 4; ++dy)
#pragma unroll
      for (int dx = 0; dx < 4; ++dx)
        s += src[(size_t)c * IMG_ * IMG_ + (py * 4 + dy) * IMG_ + (px * 4 + dx)];
    pool[c] = s * (1.f / 16.f);
  }
  for (int oc = 0; oc < RGBE_; ++oc)
    dst[(size_t)oc * NPIX + p] = w[oc*3]*pool[0] + w[oc*3+1]*pool[1] + w[oc*3+2]*pool[2] + bias[oc];
}

// =====================================================================
// K2: target token GEMM [B*Q,291] x [291->128]  (f32 WMMA 16x16x4)
// segmented k-loop: no branches inside the hot loops
// =====================================================================
__global__ void k_tok_tgt(const float* __restrict__ trf, const float* __restrict__ rgbf,
                          const float* __restrict__ tco, const float* __restrict__ tma,
                          const float* __restrict__ Wt, const float* __restrict__ bias,
                          float* __restrict__ qf, _Float16* __restrict__ qh) {
  int wave = threadIdx.x >> 5, lane = threadIdx.x & 31;
  int n = lane & 15; bool hi = lane >= 16;
  int koff = hi ? 2 : 0;
  int gq0 = blockIdx.x * 16;
  int b = gq0 / Q_, q0 = gq0 % Q_, t = q0 / NPIX;
  int img = b * T_ + t;
  int p = (q0 % NPIX) + n;                    // pixel for A-row M=n
  int oc = wave * 16 + n;
  const float* roi = trf + ((size_t)img * ROI_) * NPIX + p;     // [c]*NPIX stride
  const float* rgb = rgbf + ((size_t)img * RGBE_) * NPIX + p;
  const float* wt  = Wt + oc;                                    // [c]*128 stride
  v8f acc = {};
  for (int k0 = 0; k0 < ROI_; k0 += 4) {                 // ROI channels
    int c0 = k0 + koff;
    v2f a  = { roi[(size_t)c0 * NPIX], roi[(size_t)(c0 + 1) * NPIX] };
    v2f bb = { wt [(size_t)c0 * TOK_], wt [(size_t)(c0 + 1) * TOK_] };
    acc = wmma_f32x4(a, bb, acc);
  }
  for (int k0 = ROI_; k0 < ROI_ + RGBE_; k0 += 4) {      // RGB-embed channels
    int c0 = k0 + koff;
    v2f a  = { rgb[(size_t)(c0 - ROI_) * NPIX], rgb[(size_t)(c0 + 1 - ROI_) * NPIX] };
    v2f bb = { wt [(size_t)c0 * TOK_],          wt [(size_t)(c0 + 1) * TOK_] };
    acc = wmma_f32x4(a, bb, acc);
  }
  {                                                       // tail: coord(2)+mask(1)+pad
    v2f a, bb;
    a.x = hi ? tma[(size_t)img * NPIX + p] : tco[((size_t)img * 2)     * NPIX + p];
    a.y = hi ? 0.f                         : tco[((size_t)img * 2 + 1) * NPIX + p];
    bb.x = wt[(size_t)(288 + koff) * TOK_];
    bb.y = wt[(size_t)(289 + koff) * TOK_];
    acc = wmma_f32x4(a, bb, acc);
  }
#pragma unroll
  for (int r = 0; r < 8; ++r) {
    int q = gq0 + r + (hi ? 8 : 0);
    float v = acc[r] + bias[oc];
    qf[(size_t)q * TOK_ + oc] = v;
    qh[(size_t)q * TOK_ + oc] = (_Float16)v;
  }
}

// K3: context token GEMM [B*K,300] x [300->128]
__global__ void k_tok_ctx(const float* __restrict__ crf, const float* __restrict__ rgbf,
                          const float* __restrict__ cxyz, const float* __restrict__ cview,
                          const float* __restrict__ ccam, const float* __restrict__ cco,
                          const float* __restrict__ cma, const float* __restrict__ Wt,
                          const float* __restrict__ bias,
                          float* __restrict__ bankf, _Float16* __restrict__ bankh) {
  int wave = threadIdx.x >> 5, lane = threadIdx.x & 31;
  int n = lane & 15; bool hi = lane >= 16;
  int koff = hi ? 2 : 0;
  int gk0 = blockIdx.x * 16;
  int b = gk0 / K_, kk0 = gk0 % K_, ci = kk0 / NPIX;
  int img = b * C_ + ci;
  int p = (kk0 % NPIX) + n;
  int oc = wave * 16 + n;
  const float* roi = crf + ((size_t)img * ROI_) * NPIX + p;
  const float* rgb = rgbf + ((size_t)img * RGBE_) * NPIX + p;
  const float* wt  = Wt + oc;
  v8f acc = {};
  for (int k0 = 0; k0 < ROI_; k0 += 4) {
    int c0 = k0 + koff;
    v2f a  = { roi[(size_t)c0 * NPIX], roi[(size_t)(c0 + 1) * NPIX] };
    v2f bb = { wt [(size_t)c0 * TOK_], wt [(size_t)(c0 + 1) * TOK_] };
    acc = wmma_f32x4(a, bb, acc);
  }
  for (int k0 = ROI_; k0 < ROI_ + RGBE_; k0 += 4) {
    int c0 = k0 + koff;
    v2f a  = { rgb[(size_t)(c0 - ROI_) * NPIX], rgb[(size_t)(c0 + 1 - ROI_) * NPIX] };
    v2f bb = { wt [(size_t)c0 * TOK_],          wt [(size_t)(c0 + 1) * TOK_] };
    acc = wmma_f32x4(a, bb, acc);
  }
  // tail: xyz(3) + view(3) + cam(3) + coord(2) + mask(1) = channels 288..299
  float e0  = cxyz [((size_t)img*3    ) * NPIX + p];
  float e1  = cxyz [((size_t)img*3 + 1) * NPIX + p];
  float e2  = cxyz [((size_t)img*3 + 2) * NPIX + p];
  float e3  = cview[((size_t)img*3    ) * NPIX + p];
  float e4  = cview[((size_t)img*3 + 1) * NPIX + p];
  float e5  = cview[((size_t)img*3 + 2) * NPIX + p];
  float e6  = ccam [((size_t)img*3    ) * NPIX + p];
  float e7  = ccam [((size_t)img*3 + 1) * NPIX + p];
  float e8  = ccam [((size_t)img*3 + 2) * NPIX + p];
  float e9  = cco  [((size_t)img*2    ) * NPIX + p];
  float e10 = cco  [((size_t)img*2 + 1) * NPIX + p];
  float e11 = cma  [ (size_t)img        * NPIX + p];
  {
    v2f a = { hi ? e2 : e0, hi ? e3 : e1 };
    v2f bb = { wt[(size_t)(288 + koff) * TOK_], wt[(size_t)(289 + koff) * TOK_] };
    acc = wmma_f32x4(a, bb, acc);
  }
  {
    v2f a = { hi ? e6 : e4, hi ? e7 : e5 };
    v2f bb = { wt[(size_t)(292 + koff) * TOK_], wt[(size_t)(293 + koff) * TOK_] };
    acc = wmma_f32x4(a, bb, acc);
  }
  {
    v2f a = { hi ? e10 : e8, hi ? e11 : e9 };
    v2f bb = { wt[(size_t)(296 + koff) * TOK_], wt[(size_t)(297 + koff) * TOK_] };
    acc = wmma_f32x4(a, bb, acc);
  }
#pragma unroll
  for (int r = 0; r < 8; ++r) {
    int kk = gk0 + r + (hi ? 8 : 0);
    float v = acc[r] + bias[oc];
    bankf[(size_t)kk * TOK_ + oc] = v;
    bankh[(size_t)kk * TOK_ + oc] = (_Float16)v;
  }
}

// K4: bank_xyz / bank_conf outputs + log-conf (with validity mask)
__global__ void k_bankmeta(const float* __restrict__ cxyz, const float* __restrict__ cma,
                           const unsigned char* __restrict__ cvalid,
                           float* __restrict__ obxyz, float* __restrict__ obconf,
                           float* __restrict__ logconf) {
  int idx = blockIdx.x * 256 + threadIdx.x;          // B*K
  int b = idx / K_, kk = idx % K_, ci = kk / NPIX, p = kk % NPIX;
  int img = b * C_ + ci;
#pragma unroll
  for (int j = 0; j < 3; ++j)
    obxyz[(size_t)idx * 3 + j] = cxyz[((size_t)img * 3 + j) * NPIX + p];
  float conf = cma[(size_t)img * NPIX + p];
  obconf[idx] = conf;
  bool valid = cvalid[img] != 0;
  logconf[idx] = valid ? logf(fmaxf(conf, 1e-4f)) : -__builtin_inff();
}

// =====================================================================
// K5: attention scores (f16 WMMA 16x16x32). A-tile (16x128 halfs, 4KB)
// staged into LDS with async DMA (ASYNCcnt path), shared by all 8 waves.
// =====================================================================
__global__ void k_score(const _Float16* __restrict__ qh, const _Float16* __restrict__ bankh,
                        const float* __restrict__ logconf, float* __restrict__ outw) {
  __shared__ _Float16 sA[16 * TOK_];                 // 4 KB
  int blk = blockIdx.x;
  int b  = blk / ((Q_/16) * (K_/128));
  int r  = blk % ((Q_/16) * (K_/128));
  int qt = r / (K_/128), kt = r % (K_/128);
  // ---- async stage of the A tile: 256 threads x 16B = 4KB ----
  {
    const _Float16* gA = qh + ((size_t)b * Q_ + qt * 16) * TOK_;
    unsigned voff = (unsigned)(threadIdx.x * 16);                       // byte offset
    unsigned ldsoff = (unsigned)(uintptr_t)(&sA[0]) + voff;             // low 32b = LDS addr
    asm volatile("global_load_async_to_lds_b128 %0, %1, %2"
                 :: "v"(ldsoff), "v"(voff), "s"(gA) : "memory");
    asm volatile("s_wait_asynccnt 0x0" ::: "memory");
  }
  __syncthreads();
  int wave = threadIdx.x >> 5, lane = threadIdx.x & 31;
  int n = lane & 15; bool hi = lane >= 16;
  const _Float16* ap = &sA[(size_t)n * TOK_];                                     // A row M=n
  const _Float16* bp = bankh + ((size_t)b * K_ + kt * 128 + wave * 16 + n) * TOK_; // B col N=n
  int aoff = hi ? 8 : 0, boff = hi ? 8 : 0;
  v8f acc = {};
#pragma unroll
  for (int d0 = 0; d0 < TOK_; d0 += 32) {
    v16h a, bb;
#pragma unroll
    for (int i = 0; i < 8; ++i) { a[i] = ap[d0 + aoff + i]; a[8 + i] = ap[d0 + 16 + aoff + i]; }
#pragma unroll
    for (int i = 0; i < 8; ++i) { bb[i] = bp[d0 + boff + i]; bb[8 + i] = bp[d0 + 16 + boff + i]; }
    acc = wmma_f16(a, bb, acc);
  }
  const float scale = 0.088388347648318447f;   // 1/sqrt(128)
  int kc = kt * 128 + wave * 16 + n;
  float lc = logconf[(size_t)b * K_ + kc];
#pragma unroll
  for (int rr = 0; rr < 8; ++rr) {
    int M = rr + (hi ? 8 : 0);
    outw[((size_t)b * Q_ + qt * 16 + M) * K_ + kc] = acc[rr] * scale + lc;
  }
}

// K6: per-row online max/sum-of-exp over K=16384
__global__ void k_rowstats(const float* __restrict__ w, float* __restrict__ rmax,
                           float* __restrict__ rsum) {
  __shared__ float sm[256], ss[256];
  int row = blockIdx.x;
  const float* wr = w + (size_t)row * K_;
  float m = -__builtin_inff(), s = 0.f;
  for (int k = threadIdx.x; k < K_; k += 256) {
    float v = wr[k];
    if (v == -__builtin_inff()) continue;
    if (v > m) { s = s * __expf(m - v) + 1.f; m = v; }
    else       { s += __expf(v - m); }
  }
  sm[threadIdx.x] = m; ss[threadIdx.x] = s;
  __syncthreads();
  for (int st = 128; st > 0; st >>= 1) {
    if (threadIdx.x < st) {
      float m2 = sm[threadIdx.x + st], s2 = ss[threadIdx.x + st];
      float M = fmaxf(sm[threadIdx.x], m2);
      float S;
      if (M == -__builtin_inff()) S = 0.f;
      else S = ss[threadIdx.x] * __expf(sm[threadIdx.x] - M) + s2 * __expf(m2 - M);
      sm[threadIdx.x] = M; ss[threadIdx.x] = S;
    }
    __syncthreads();
  }
  if (threadIdx.x == 0) { rmax[row] = sm[0]; rsum[row] = ss[0]; }
}

// K7: normalize weights in place + xyz first/second moments + amb8
__global__ void k_norm(float* __restrict__ w, const float* __restrict__ rmax,
                       const float* __restrict__ rsum, const float* __restrict__ bxyz,
                       float* __restrict__ xyz_e, float* __restrict__ xyz_2,
                       float* __restrict__ amb8) {
  __shared__ float red[6][256];
  int row = blockIdx.x;
  int b = row / Q_;
  float m = rmax[row], s = rsum[row];
  float inv = (s > 0.f && m > -3.0e38f) ? 1.f / s : 0.f;
  float* wr = w + (size_t)row * K_;
  const float* bx = bxyz + (size_t)b * K_ * 3;
  float e[3] = {0,0,0}, f[3] = {0,0,0};
  for (int k = threadIdx.x; k < K_; k += 256) {
    float v = wr[k];
    float ww = (inv == 0.f) ? 0.f : __expf(v - m) * inv;
    wr[k] = ww;
    float x0 = bx[(size_t)k*3], x1 = bx[(size_t)k*3+1], x2 = bx[(size_t)k*3+2];
    e[0] += ww*x0; e[1] += ww*x1; e[2] += ww*x2;
    f[0] += ww*x0*x0; f[1] += ww*x1*x1; f[2] += ww*x2*x2;
  }
#pragma unroll
  for (int j = 0; j < 3; ++j) { red[j][threadIdx.x] = e[j]; red[3+j][threadIdx.x] = f[j]; }
  __syncthreads();
  for (int st = 128; st > 0; st >>= 1) {
    if (threadIdx.x < st)
#pragma unroll
      for (int j = 0; j < 6; ++j) red[j][threadIdx.x] += red[j][threadIdx.x + st];
    __syncthreads();
  }
  if (threadIdx.x == 0) {
    float var = 0.f;
#pragma unroll
    for (int j = 0; j < 3; ++j) {
      float E = red[j][0], F = red[3+j][0];
      xyz_e[(size_t)row*3 + j] = E;
      xyz_2[(size_t)row*3 + j] = F;
      var += fmaxf(F - E * E, 0.f);
    }
    amb8[row] = sqrtf(var * (1.f/3.f) + 1e-6f);
  }
}

// =====================================================================
// K8: feat_ret = weights x bank_feat (f32 WMMA 16x16x4)
// wave = 16q x 32d with two independent accumulator chains + prefetch
// =====================================================================
__global__ void k_featret(const float* __restrict__ w, const float* __restrict__ bankf,
                          float* __restrict__ fret) {
  int wave = threadIdx.x >> 5, lane = threadIdx.x & 31;
  int n = lane & 15; bool hi = lane >= 16;
  int koff = hi ? 2 : 0;
  int wg = blockIdx.x * 8 + wave;                 // 1024 waves total
  int b = wg / ((Q_/16) * (TOK_/32));
  int r = wg % ((Q_/16) * (TOK_/32));
  int qt = r / (TOK_/32), dt2 = r % (TOK_/32);
  const float* ap = w + ((size_t)b * Q_ + qt * 16 + n) * K_;
  const float* bb0 = bankf + (size_t)b * K_ * TOK_ + dt2 * 32 + n;
  v8f acc0 = {}, acc1 = {};
  for (int k0 = 0; k0 < K_; k0 += 4) {
    if ((k0 & 255) == 0) __builtin_prefetch(ap + k0 + 1024, 0, 3);  // global_prefetch
    v2f a = { ap[k0 + koff], ap[k0 + koff + 1] };
    const float* bp = bb0 + (size_t)(k0 + koff) * TOK_;
    v2f b0 = { bp[0],  bp[TOK_] };
    v2f b1 = { bp[16], bp[16 + TOK_] };
    acc0 = wmma_f32x4(a, b0, acc0);
    acc1 = wmma_f32x4(a, b1, acc1);
  }
#pragma unroll
  for (int rr = 0; rr < 8; ++rr) {
    int M = rr + (hi ? 8 : 0);
    size_t o = ((size_t)b * Q_ + qt * 16 + M) * TOK_ + dt2 * 32 + n;
    fret[o]      = acc0[rr];
    fret[o + 16] = acc1[rr];
  }
}

// K9: retrieved = (tok + feat_ret) x ret_w^T -> out[b,t,256,16,16]
__global__ void k_retr(const float* __restrict__ qf, const float* __restrict__ fret,
                       const float* __restrict__ Wt, const float* __restrict__ bias,
                       float* __restrict__ out0) {
  int wave = threadIdx.x >> 5, lane = threadIdx.x & 31;
  int n = lane & 15; bool hi = lane >= 16;
  int koff = hi ? 2 : 0;
  int wg = blockIdx.x * 8 + wave;              // 4096 waves
  int qt = wg / (ROI_/16), oct = wg % (ROI_/16);
  int arow = qt * 16 + n;
  int oc = oct * 16 + n;
  const float* wt = Wt + oc;                   // [k]*256 stride
  v8f acc = {};
  for (int k0 = 0; k0 < TOK_; k0 += 4) {
    size_t ai = (size_t)arow * TOK_ + k0 + koff;
    v2f a  = { qf[ai] + fret[ai], qf[ai + 1] + fret[ai + 1] };
    v2f bb = { wt[(size_t)(k0 + koff) * ROI_], wt[(size_t)(k0 + koff + 1) * ROI_] };
    acc = wmma_f32x4(a, bb, acc);
  }
#pragma unroll
  for (int rr = 0; rr < 8; ++rr) {
    int row = qt * 16 + rr + (hi ? 8 : 0);
    int b = row / Q_, t = (row % Q_) / NPIX, p = row % NPIX;
    out0[(((size_t)(b * T_ + t)) * ROI_ + oc) * NPIX + p] = acc[rr] + bias[oc];
  }
}

// =====================================================================
// K10: 3x3 conv on prior_in (261ch) via tap-major im2col WMMA + GELU
// =====================================================================
__global__ void k_conv3(const float* __restrict__ qf, const float* __restrict__ fret,
                        const float* __restrict__ xe, const float* __restrict__ amb8,
                        const float* __restrict__ Wt, const float* __restrict__ bias,
                        float* __restrict__ h1) {
  int wave = threadIdx.x >> 5, lane = threadIdx.x & 31;
  int n = lane & 15; bool hi = lane >= 16;
  int koff = hi ? 2 : 0;
  int wg = blockIdx.x * 8 + wave;              // 2048 waves
  int qt = wg / (TOK_/16), oct = wg % (TOK_/16);
  int gr = qt * 16 + n;                         // this lane's A-row pixel
  int b = gr / Q_, tq = gr % Q_, t = tq / NPIX, p = tq % NPIX;
  int y = p >> 4, x = p & 15;
  int baseq = b * Q_ + t * NPIX;
  int oc = oct * 16 + n;
  v8f acc = {};
  for (int t9 = 0; t9 < 9; ++t9) {
    int dy = t9 / 3 - 1, dx = t9 % 3 - 1;
    int yy = y + dy, xx = x + dx;
    bool ok = ((unsigned)yy < (unsigned)R_) && ((unsigned)xx < (unsigned)R_);
    int qidx = ok ? (baseq + yy * R_ + xx) : baseq;
    const float* aq = qf   + (size_t)qidx * TOK_;
    const float* af = fret + (size_t)qidx * TOK_;
    const float* wt = Wt + (size_t)t9 * PRINP * TOK_ + oc;
    for (int k0 = 0; k0 < TOK_; k0 += 4) {                // tgt_tok channels
      int c0 = k0 + koff;
      v2f a  = { ok ? aq[c0] : 0.f, ok ? aq[c0 + 1] : 0.f };
      v2f bb = { wt[(size_t)c0 * TOK_], wt[(size_t)(c0 + 1) * TOK_] };
      acc = wmma_f32x4(a, bb, acc);
    }
    for (int k0 = TOK_; k0 < 2 * TOK_; k0 += 4) {         // feat_ret channels
      int c0 = k0 + koff;
      v2f a  = { ok ? af[c0 - TOK_] : 0.f, ok ? af[c0 + 1 - TOK_] : 0.f };
      v2f bb = { wt[(size_t)c0 * TOK_], wt[(size_t)(c0 + 1) * TOK_] };
      acc = wmma_f32x4(a, bb, acc);
    }
    float xe0 = 0.f, xe1 = 0.f, xe2 = 0.f, a8v = 0.f, c8v = 0.f;
    if (ok) {
      xe0 = xe[(size_t)qidx * 3];
      xe1 = xe[(size_t)qidx * 3 + 1];
      xe2 = xe[(size_t)qidx * 3 + 2];
      a8v = amb8[qidx];
      c8v = expf(-a8v);
    }
    {   // channels 256..259 : xyz_ret(3) + conf8
      v2f a  = { hi ? xe2 : xe0, hi ? c8v : xe1 };
      v2f bb = { wt[(size_t)(256 + koff) * TOK_], wt[(size_t)(257 + koff) * TOK_] };
      acc = wmma_f32x4(a, bb, acc);
    }
    {   // channels 260..263 : amb8 + zero pad
      v2f a  = { hi ? 0.f : a8v, 0.f };
      v2f bb = { wt[(size_t)(260 + koff) * TOK_], wt[(size_t)(261 + koff) * TOK_] };
      acc = wmma_f32x4(a, bb, acc);
    }
  }
#pragma unroll
  for (int rr = 0; rr < 8; ++rr) {
    int grow = qt * 16 + rr + (hi ? 8 : 0);
    float h = acc[rr] + bias[oc];
    float g = 0.5f * h * (1.f + erff(h * 0.7071067811865475f));   // exact GELU
    h1[(size_t)grow * TOK_ + oc] = g;
  }
}

// K11: 1x1 head (128 -> 5) at 16x16 (commutes with bilinear upsample)
__global__ void k_head(const float* __restrict__ h1, const float* __restrict__ W,
                       const float* __restrict__ bias, float* __restrict__ praw) {
  int bt = blockIdx.x, p = threadIdx.x;
  const float* hp = h1 + ((size_t)bt * NPIX + p) * TOK_;
  float acc[5];
#pragma unroll
  for (int j = 0; j < 5; ++j) acc[j] = bias[j];
  for (int d = 0; d < TOK_; ++d) {
    float hv = hp[d];
#pragma unroll
    for (int j = 0; j < 5; ++j) acc[j] += W[j * TOK_ + d] * hv;
  }
#pragma unroll
  for (int j = 0; j < 5; ++j) praw[((size_t)bt * 5 + j) * NPIX + p] = acc[j];
}

// K12: bilinear 16->64 upsample + sigmoid heads -> xyz_prior / conf / ambiguity
__device__ inline float bilerp(const float* base, int y0, int y1, int x0, int x1,
                               float fy, float fx) {
  float v00 = base[y0*R_+x0], v01 = base[y0*R_+x1];
  float v10 = base[y1*R_+x0], v11 = base[y1*R_+x1];
  return (v00*(1.f-fx)+v01*fx)*(1.f-fy) + (v10*(1.f-fx)+v11*fx)*fy;
}
__global__ void k_up(const float* __restrict__ praw, const float* __restrict__ amb8,
                     float* __restrict__ oxyz, float* __restrict__ oconf,
                     float* __restrict__ oamb) {
  int idx = blockIdx.x * 256 + threadIdx.x;    // B*T*64*64
  int bt = idx >> 12, pp = idx & 4095, y = pp >> 6, x = pp & 63;
  float sy = fmaxf((y + 0.5f) * 0.25f - 0.5f, 0.f);
  float sx = fmaxf((x + 0.5f) * 0.25f - 0.5f, 0.f);
  int y0 = (int)sy, x0 = (int)sx;
  int y1 = min(y0 + 1, R_ - 1), x1 = min(x0 + 1, R_ - 1);
  float fy = sy - (float)y0, fx = sx - (float)x0;
  float raw[5];
#pragma unroll
  for (int j = 0; j < 5; ++j)
    raw[j] = bilerp(praw + ((size_t)bt * 5 + j) * NPIX, y0, y1, x0, x1, fy, fx);
  float seed = bilerp(amb8 + (size_t)bt * NPIX, y0, y1, x0, x1, fy, fx);
#pragma unroll
  for (int j = 0; j < 3; ++j) oxyz[((size_t)bt * 3 + j) * 4096 + pp] = raw[j];
  float amb = (1.f / (1.f + expf(-raw[3]))) * seed;
  float cf  = (1.f / (1.f + expf(-raw[4]))) * expf(-amb);
  oconf[(size_t)bt * 4096 + pp] = fminf(fmaxf(cf, 0.f), 1.f);
  oamb [(size_t)bt * 4096 + pp] = fminf(fmaxf(amb, 0.f), 1.f);
}

} // namespace

extern "C" void kernel_launch(void* const* d_in, const int* in_sizes, int n_in,
                              void* d_out, int out_size, void* d_ws, size_t ws_size,
                              hipStream_t stream) {
  const float* trf   = (const float*)d_in[0];
  const float* crf   = (const float*)d_in[1];
  const float* cxyz  = (const float*)d_in[2];
  const float* cview = (const float*)d_in[3];
  const float* ccam  = (const float*)d_in[4];
  const float* trgb  = (const float*)d_in[5];
  const float* crgb  = (const float*)d_in[6];
  const float* tco   = (const float*)d_in[7];
  const float* cco   = (const float*)d_in[8];
  const float* tma   = (const float*)d_in[9];
  const float* cma   = (const float*)d_in[10];
  const unsigned char* cvalid = (const unsigned char*)d_in[12];
  const float* rgb_w = (const float*)d_in[13];
  const float* rgb_b = (const float*)d_in[14];
  const float* tgt_w = (const float*)d_in[15];
  const float* tgt_b = (const float*)d_in[16];
  const float* ctx_w = (const float*)d_in[17];
  const float* ctx_b = (const float*)d_in[18];
  const float* ret_w = (const float*)d_in[19];
  const float* ret_b = (const float*)d_in[20];
  const float* p1_w  = (const float*)d_in[21];
  const float* p1_b  = (const float*)d_in[22];
  const float* p2_w  = (const float*)d_in[23];
  const float* p2_b  = (const float*)d_in[24];

  float* out = (float*)d_out;
  float* out_ret  = out + OUT_RET;
  float* out_xyz  = out + OUT_XYZ;
  float* out_conf = out + OUT_CONF;
  float* out_amb  = out + OUT_AMB;
  float* out_w    = out + OUT_W;
  float* out_bxyz = out + OUT_BXYZ;
  float* out_bconf= out + OUT_BCONF;

  // ---- workspace carve-out (floats) ----
  float* ws = (float*)d_ws;
  float* tgt_rgbf = ws;                                      // 16*32*256
  float* ctx_rgbf = tgt_rgbf + (size_t)B_*T_*RGBE_*NPIX;     // 128*32*256
  float* qf       = ctx_rgbf + (size_t)B_*C_*RGBE_*NPIX;     // B*Q*128
  float* bankf    = qf       + (size_t)B_*Q_*TOK_;           // B*K*128
  float* logconf  = bankf    + (size_t)B_*K_*TOK_;           // B*K
  float* rowmax   = logconf  + (size_t)B_*K_;                // B*Q
  float* rowsum   = rowmax   + (size_t)B_*Q_;
  float* xyz_e    = rowsum   + (size_t)B_*Q_;                // B*Q*3
  float* xyz_2    = xyz_e    + (size_t)B_*Q_*3;
  float* amb8     = xyz_2    + (size_t)B_*Q_*3;              // B*Q
  float* fret     = amb8     + (size_t)B_*Q_;                // B*Q*128
  float* h1       = fret     + (size_t)B_*Q_*TOK_;           // B*Q*128
  float* praw     = h1       + (size_t)B_*Q_*TOK_;           // 16*5*256
  float* tgt_wt   = praw     + (size_t)B_*T_*5*NPIX;         // 292*128
  float* ctx_wt   = tgt_wt   + (size_t)292*TOK_;             // 300*128
  float* ret_wt   = ctx_wt   + (size_t)CIN_C*TOK_;           // 128*256
  float* p1_wt    = ret_wt   + (size_t)TOK_*ROI_;            // 9*264*128
  _Float16* qh    = (_Float16*)(p1_wt + (size_t)9*PRINP*TOK_);
  _Float16* bankh = qh + (size_t)B_*Q_*TOK_;

  // 0. weight transposes (lane-coalesced B operands)
  k_wprep<<<dim3((9*PRINP*TOK_)/256), dim3(256), 0, stream>>>(tgt_w, ctx_w, ret_w, p1_w,
                                                              tgt_wt, ctx_wt, ret_wt, p1_wt);
  // 1. RGB pool+conv (targets + contexts)
  k_rgbfeat<<<dim3(B_*T_ + B_*C_), dim3(256), 0, stream>>>(trgb, crgb, rgb_w, rgb_b,
                                                           tgt_rgbf, ctx_rgbf);
  // 2. token projections (f32 WMMA)
  k_tok_tgt<<<dim3(B_*Q_/16), dim3(256), 0, stream>>>(trf, tgt_rgbf, tco, tma,
                                                      tgt_wt, tgt_b, qf, qh);
  k_tok_ctx<<<dim3(B_*K_/16), dim3(256), 0, stream>>>(crf, ctx_rgbf, cxyz, cview, ccam,
                                                      cco, cma, ctx_wt, ctx_b, bankf, bankh);
  // 3. bank metadata outputs
  k_bankmeta<<<dim3(B_*K_/256), dim3(256), 0, stream>>>(cxyz, cma, cvalid,
                                                        out_bxyz, out_bconf, logconf);
  // 4. attention scores (f16 WMMA + async-to-LDS A staging)
  k_score<<<dim3(B_*(Q_/16)*(K_/128)), dim3(256), 0, stream>>>(qh, bankh, logconf, out_w);
  // 5. softmax stats + normalize (+ xyz moments)
  k_rowstats<<<dim3(B_*Q_), dim3(256), 0, stream>>>(out_w, rowmax, rowsum);
  k_norm<<<dim3(B_*Q_), dim3(256), 0, stream>>>(out_w, rowmax, rowsum, out_bxyz,
                                                xyz_e, xyz_2, amb8);
  // 6. feat_ret = weights x bank (f32 WMMA, dual acc, prefetch)
  k_featret<<<dim3(B_*(Q_/16)*(TOK_/32)/8), dim3(256), 0, stream>>>(out_w, bankf, fret);
  // 7. retrieved output (f32 WMMA)
  k_retr<<<dim3(B_*(Q_/16)*(ROI_/16)/8), dim3(256), 0, stream>>>(qf, fret, ret_wt, ret_b,
                                                                 out_ret);
  // 8. 3x3 conv + GELU (tap-major im2col f32 WMMA)
  k_conv3<<<dim3(B_*(Q_/16)*(TOK_/16)/8), dim3(256), 0, stream>>>(qf, fret, xyz_e, amb8,
                                                                  p1_wt, p1_b, h1);
  // 9. 5-channel head at 16x16, then bilinear upsample + sigmoids
  k_head<<<dim3(B_*T_), dim3(256), 0, stream>>>(h1, p2_w, p2_b, praw);
  k_up<<<dim3(B_*T_*P_*P_/256), dim3(256), 0, stream>>>(praw, amb8,
                                                        out_xyz, out_conf, out_amb);
  (void)in_sizes; (void)n_in; (void)out_size; (void)ws_size;
}
